// HyperAttention_42803644072598
// MI455X (gfx1250) — matrix operand order repaired and checked by reading, not compiled, not run on hardware
//
#include <hip/hip_runtime.h>
#include <stdint.h>

typedef __attribute__((ext_vector_type(16))) _Float16 v16h;
typedef __attribute__((ext_vector_type(8)))  float    v8f;

#define N_TOK   8192
#define D_HEAD  64
#define NB      32      // blocks of 256 per slice
#define BLK     256
#define NSLICE  32      // b*h
#define NPROJ   7
#define SAMPLE  256
#define SCALE   0.125f  // d^-0.5, d=64
#define NEGINF  (-3.0e38f)
#define LOG32   3.4657359027997265f

// ---------------------------------------------------------------------------
// Kernel 1: LSH hash (Gray code of sign bits of 7 projections)
// ---------------------------------------------------------------------------
__global__ __launch_bounds__(256) void lsh_hash_kernel(
    const float* __restrict__ q, const float* __restrict__ k,
    const float* __restrict__ proj, int* __restrict__ hq, int* __restrict__ hk)
{
  __shared__ float sp[D_HEAD * NPROJ];
  int tid = threadIdx.x;
  for (int i = tid; i < D_HEAD * NPROJ; i += 256) sp[i] = proj[i];
  __syncthreads();

  int rid   = blockIdx.x * 256 + tid;          // 0 .. 2*NSLICE*N_TOK-1
  int total = NSLICE * N_TOK;
  int which = rid >= total;
  int row   = rid - which * total;
  const float* src = (which ? k : q) + (size_t)row * D_HEAD;

  float acc[NPROJ];
#pragma unroll
  for (int r = 0; r < NPROJ; ++r) acc[r] = 0.f;
  for (int d = 0; d < D_HEAD; ++d) {
    float x = src[d];
#pragma unroll
    for (int r = 0; r < NPROJ; ++r) acc[r] += x * sp[d * NPROJ + r];
  }
  int code = 0;
#pragma unroll
  for (int r = 0; r < NPROJ; ++r) code |= (acc[r] > 0.f) ? (1 << r) : 0;
  int bucket = code ^ (code >> 1);             // reflected Gray code
  (which ? hk : hq)[row] = bucket;
}

// ---------------------------------------------------------------------------
// Kernel 2: stable counting sort by bucket (128 buckets), one WG per slice/side
// Emits idx such that idx[pos] = original row (== stable argsort of hashes).
// ---------------------------------------------------------------------------
__global__ __launch_bounds__(256) void stable_bucket_sort_kernel(
    const int* __restrict__ hq, const int* __restrict__ hk,
    int* __restrict__ qidx, int* __restrict__ kidx)
{
  __shared__ int hist[128];
  __shared__ int running[128];
  __shared__ int wavehist[8][128];

  int tid  = threadIdx.x;
  int lane = tid & 31;
  int wv   = tid >> 5;
  int sel  = blockIdx.x >> 5;                  // 0 = q, 1 = k
  int slice = blockIdx.x & 31;
  const int* hsh  = (sel ? hk : hq) + slice * N_TOK;
  int*       oidx = (sel ? kidx : qidx) + slice * N_TOK;

  for (int i = tid; i < 128; i += 256) hist[i] = 0;
  __syncthreads();
  for (int i = tid; i < N_TOK; i += 256) atomicAdd(&hist[hsh[i]], 1);
  __syncthreads();
  if (tid == 0) {
    int acc = 0;
    for (int v = 0; v < 128; ++v) { running[v] = acc; acc += hist[v]; }
  }
  __syncthreads();

  unsigned lowmask = (1u << lane) - 1u;
  int* wh = &wavehist[0][0];
  for (int c = 0; c < N_TOK / 256; ++c) {
    int i = c * 256 + tid;
    int v = hsh[i];
#pragma unroll
    for (int z = 0; z < 4; ++z) wh[tid + z * 256] = 0;
    __syncthreads();
    unsigned mm = 0xffffffffu;                 // lanes in wave with same bucket
#pragma unroll
    for (int b = 0; b < NPROJ; ++b) {
      unsigned bal = (unsigned)__ballot((v >> b) & 1);
      mm &= ((v >> b) & 1) ? bal : ~bal;
    }
    int lrank = __popc(mm & lowmask);
    if ((mm & lowmask) == 0u) wavehist[wv][v] = __popc(mm);
    __syncthreads();
    int before = 0;
    for (int w = 0; w < 8; ++w) if (w < wv) before += wavehist[w][v];
    oidx[running[v] + before + lrank] = i;
    __syncthreads();
    if (tid < 128) {
      int s = 0;
#pragma unroll
      for (int w = 0; w < 8; ++w) s += wavehist[w][tid];
      running[tid] += s;
    }
    __syncthreads();
  }
}

// ---------------------------------------------------------------------------
// Kernel 3: fused block-diagonal attention + sampled residual + LSE-combine +
//           un-sort scatter. One WG per (block g, slice). 16 waves, 1 strip of
//           16 sorted queries each. Computes S^T = K * (Q*scale)^T so softmax
//           reductions are in-lane and P^T feeds WMMA B-fragments via shuffles.
// ---------------------------------------------------------------------------
struct Blk8 { int v[8]; };

__global__ __launch_bounds__(512) void hyper_attn_kernel(
    const float* __restrict__ query, const float* __restrict__ key,
    const float* __restrict__ value, const int* __restrict__ sampled,
    const int* __restrict__ qidx, const int* __restrict__ kidx,
    float* __restrict__ out)
{
  __shared__ _Float16 sK [BLK * D_HEAD];       // row-major [key][d]   (32 KB)
  __shared__ _Float16 sVT[D_HEAD * BLK];       // transposed [d][key]  (32 KB)

  const int g     = blockIdx.x;
  const int slice = blockIdx.y;
  const int tid   = threadIdx.x;
  const int lane  = tid & 31;
  const int hh    = (lane >> 4) & 1;           // lane half
  const int m     = lane & 15;                 // query column within strip
  const int wv    = tid >> 5;                  // strip id (16 rows each)

  const size_t sbase = (size_t)slice * N_TOK;
  const float* Qs = query + sbase * D_HEAD;
  const float* Ks = key   + sbase * D_HEAD;
  const float* Vs = value + sbase * D_HEAD;
  const int*   qI = qidx + sbase;
  const int*   kI = kidx + sbase;

  // ---- gather block-g sorted K (row-major) and V (transposed) to LDS ----
  {
    int r   = tid & 255;
    int isv = tid >> 8;
    int kr  = kI[g * BLK + r];
    const float* src = (isv ? Vs : Ks) + (size_t)kr * D_HEAD;
    if (!isv) {
#pragma unroll
      for (int d = 0; d < D_HEAD; d += 4) {
        float4 f = *(const float4*)(src + d);
        sK[r * D_HEAD + d + 0] = (_Float16)f.x;
        sK[r * D_HEAD + d + 1] = (_Float16)f.y;
        sK[r * D_HEAD + d + 2] = (_Float16)f.z;
        sK[r * D_HEAD + d + 3] = (_Float16)f.w;
      }
    } else {
#pragma unroll
      for (int d = 0; d < D_HEAD; d += 4) {
        float4 f = *(const float4*)(src + d);
        sVT[(d + 0) * BLK + r] = (_Float16)f.x;
        sVT[(d + 1) * BLK + r] = (_Float16)f.y;
        sVT[(d + 2) * BLK + r] = (_Float16)f.z;
        sVT[(d + 3) * BLK + r] = (_Float16)f.w;
      }
    }
  }

  // ---- per-lane Q^T B-fragments (pre-scaled), reused by both phases ----
  const int qi = qI[g * BLK + wv * 16 + m];    // original row for column m
  const float* qrow = Qs + (size_t)qi * D_HEAD;
  v16h bq0, bq1;
#pragma unroll
  for (int j = 0; j < 8; ++j) {
    int d0 = (j >> 2) * 16 + hh * 8 + (j & 3) * 2; // B-layout: K = (j/4)*16 + (l/16)*8 + 2(j%4)
    float2 f0 = *(const float2*)(qrow + d0);
    bq0[2 * j]     = (_Float16)(f0.x * SCALE);
    bq0[2 * j + 1] = (_Float16)(f0.y * SCALE);
    float2 f1 = *(const float2*)(qrow + 32 + d0);
    bq1[2 * j]     = (_Float16)(f1.x * SCALE);
    bq1[2 * j + 1] = (_Float16)(f1.y * SCALE);
  }
  __syncthreads();

  Blk8 blk;                                    // sampled-column block ids (phase 2)
#pragma unroll
  for (int i = 0; i < 8; ++i) blk.v[i] = 0;

  auto run_phase = [&](bool masked, float& lse, v8f O[4]) {
    v8f S[16];                                 // S^T tiles: row=key, col=query
#pragma unroll
    for (int t = 0; t < 16; ++t) {
      v16h a0, a1;                             // A = K rows t*16+m, d-chunks
#pragma unroll
      for (int j = 0; j < 8; ++j) {
        int kk = (j >> 2) * 16 + hh * 8 + (j & 3) * 2;
        const _Float16* p = &sK[(t * 16 + m) * D_HEAD + kk];
        a0[2 * j] = p[0];  a0[2 * j + 1] = p[1];
        a1[2 * j] = p[32]; a1[2 * j + 1] = p[33];
      }
      v8f c = {0.f, 0.f, 0.f, 0.f, 0.f, 0.f, 0.f, 0.f};
      c = __builtin_amdgcn_wmma_f32_16x16x32_f16(false, a0, false, bq0, (short)0, c, false, false);
      c = __builtin_amdgcn_wmma_f32_16x16x32_f16(false, a1, false, bq1, (short)0, c, false, false);
      S[t] = c;
    }
    if (masked) {                              // key's sampled block == g -> -inf
#pragma unroll
      for (int t = 0; t < 16; ++t)
#pragma unroll
        for (int r = 0; r < 8; ++r) {
          int bval = __shfl(blk.v[r], 2 * t + hh, 32);
          if (bval == g) S[t][r] = NEGINF;
        }
    }
    float mx = NEGINF;
#pragma unroll
    for (int t = 0; t < 16; ++t)
#pragma unroll
      for (int r = 0; r < 8; ++r) mx = fmaxf(mx, S[t][r]);
    mx = fmaxf(mx, __shfl_xor(mx, 16, 32));
    float lsum = 0.f;
#pragma unroll
    for (int t = 0; t < 16; ++t)
#pragma unroll
      for (int r = 0; r < 8; ++r) {
        float p = __expf(S[t][r] - mx);
        S[t][r] = p;
        lsum += p;
      }
    lsum += __shfl_xor(lsum, 16, 32);
    lse = mx + __logf(lsum);

#pragma unroll
    for (int dt = 0; dt < 4; ++dt) O[dt] = (v8f){0.f,0.f,0.f,0.f,0.f,0.f,0.f,0.f};
#pragma unroll
    for (int kc = 0; kc < 8; ++kc) {           // O^T = V^T * P^T, 32-key chunks
      v16h bp;                                 // P^T B-fragment from S^T regs
#pragma unroll
      for (int j = 0; j < 8; ++j) {
        float v0, v1;
        if (j < 4) {
          float o0 = __shfl_xor(S[2 * kc + 1][2 * j],     16, 32);
          float o1 = __shfl_xor(S[2 * kc + 1][2 * j + 1], 16, 32);
          v0 = hh ? o0 : S[2 * kc][2 * j];
          v1 = hh ? o1 : S[2 * kc][2 * j + 1];
        } else {
          float o0 = __shfl_xor(S[2 * kc][2 * j - 8], 16, 32);
          float o1 = __shfl_xor(S[2 * kc][2 * j - 7], 16, 32);
          v0 = hh ? S[2 * kc + 1][2 * j - 8] : o0;
          v1 = hh ? S[2 * kc + 1][2 * j - 7] : o1;
        }
        bp[2 * j] = (_Float16)v0; bp[2 * j + 1] = (_Float16)v1;
      }
#pragma unroll
      for (int dt = 0; dt < 4; ++dt) {
        v16h av;                               // A = V^T[d-tile][key-chunk]
#pragma unroll
        for (int j = 0; j < 8; ++j) {
          int kk = kc * 32 + hh * 16 + 2 * j;
          const _Float16* p = &sVT[(dt * 16 + m) * BLK + kk];
          av[2 * j] = p[0]; av[2 * j + 1] = p[1];
        }
        O[dt] = __builtin_amdgcn_wmma_f32_16x16x32_f16(false, av, false, bp, (short)0, O[dt], false, false);
      }
    }
  };

  float lse1, lse2;
  v8f O1[4], O2[4];
  run_phase(false, lse1, O1);                  // block-diagonal exact attention
  __syncthreads();

  // ---- regather: sampled sorted K/V columns into same LDS ----
  {
    int r   = tid & 255;
    int isv = tid >> 8;
    int ss  = sampled[slice * SAMPLE + r];
    int kr  = kI[ss];
    const float* src = (isv ? Vs : Ks) + (size_t)kr * D_HEAD;
    if (!isv) {
#pragma unroll
      for (int d = 0; d < D_HEAD; d += 4) {
        float4 f = *(const float4*)(src + d);
        sK[r * D_HEAD + d + 0] = (_Float16)f.x;
        sK[r * D_HEAD + d + 1] = (_Float16)f.y;
        sK[r * D_HEAD + d + 2] = (_Float16)f.z;
        sK[r * D_HEAD + d + 3] = (_Float16)f.w;
      }
    } else {
#pragma unroll
      for (int d = 0; d < D_HEAD; d += 4) {
        float4 f = *(const float4*)(src + d);
        sVT[(d + 0) * BLK + r] = (_Float16)f.x;
        sVT[(d + 1) * BLK + r] = (_Float16)f.y;
        sVT[(d + 2) * BLK + r] = (_Float16)f.z;
        sVT[(d + 3) * BLK + r] = (_Float16)f.w;
      }
    }
  }
#pragma unroll
  for (int i = 0; i < 8; ++i)
    blk.v[i] = sampled[slice * SAMPLE + lane * 8 + i] >> 8;   // //256
  __syncthreads();

  run_phase(true, lse2, O2);                   // sampled residual attention
  lse2 += LOG32;                               // importance weight log(n/S)

  // ---- combine and scatter (un-sort) ----
  float cw = 1.f / (1.f + __expf(lse2 - lse1));
  float* orow = out + (sbase + (size_t)qi) * D_HEAD;
#pragma unroll
  for (int dt = 0; dt < 4; ++dt) {
    float4 f0, f1;
    f0.x = cw * O1[dt][0] + (1.f - cw) * O2[dt][0];
    f0.y = cw * O1[dt][1] + (1.f - cw) * O2[dt][1];
    f0.z = cw * O1[dt][2] + (1.f - cw) * O2[dt][2];
    f0.w = cw * O1[dt][3] + (1.f - cw) * O2[dt][3];
    f1.x = cw * O1[dt][4] + (1.f - cw) * O2[dt][4];
    f1.y = cw * O1[dt][5] + (1.f - cw) * O2[dt][5];
    f1.z = cw * O1[dt][6] + (1.f - cw) * O2[dt][6];
    f1.w = cw * O1[dt][7] + (1.f - cw) * O2[dt][7];
    int d0 = dt * 16 + hh * 8;
    *(float4*)(orow + d0)     = f0;
    *(float4*)(orow + d0 + 4) = f1;
  }
}

// ---------------------------------------------------------------------------
extern "C" void kernel_launch(void* const* d_in, const int* in_sizes, int n_in,
                              void* d_out, int out_size, void* d_ws, size_t ws_size,
                              hipStream_t stream) {
  const float* q       = (const float*)d_in[0];
  const float* k       = (const float*)d_in[1];
  const float* v       = (const float*)d_in[2];
  const float* proj    = (const float*)d_in[3];
  const int*   sampled = (const int*)d_in[4];
  float*       out     = (float*)d_out;

  int* hq   = (int*)d_ws;                       // 1 MB each
  int* hk   = hq + NSLICE * N_TOK;
  int* qidx = hk + NSLICE * N_TOK;
  int* kidx = qidx + NSLICE * N_TOK;

  lsh_hash_kernel<<<(2 * NSLICE * N_TOK) / 256, 256, 0, stream>>>(q, k, proj, hq, hk);
  stable_bucket_sort_kernel<<<2 * NSLICE, 256, 0, stream>>>(hq, hk, qidx, kidx);
  hyper_attn_kernel<<<dim3(NB, NSLICE), 512, 0, stream>>>(q, k, v, sampled, qidx, kidx, out);
}
